// PRL_Net_6777458393872
// MI455X (gfx1250) — compile-verified, matrix operand order checked by reading, not run on hardware
//
#include <hip/hip_runtime.h>

#define N_S 256
#define T_S 70
#define V_S 64
#define B_S 6
#define H_S 1024
#define C_S 4

typedef unsigned short u16;
typedef __attribute__((ext_vector_type(16))) __bf16 v16bf;
typedef __attribute__((ext_vector_type(8)))  float  v8f;
typedef __attribute__((ext_vector_type(4)))  unsigned int v4u;
typedef __attribute__((ext_vector_type(8)))  int v8i;
typedef __attribute__((ext_vector_type(4)))  int v4i;

__device__ __forceinline__ u16 f32_to_bf16(float f) {
    unsigned int u = __float_as_uint(f);
    u += 0x7FFFu + ((u >> 16) & 1u);   // round-to-nearest-even
    return (u16)(u >> 16);
}

// Branch-free tanh: 1 - 2/(exp(2x)+1) via v_exp_f32 + v_rcp_f32.
// Saturates correctly to +/-1 for large |x|, no EXEC divergence.
__device__ __forceinline__ float tanh_fast(float x) {
    float e = __builtin_amdgcn_exp2f(x * 2.88539008177792681472f); // 2*log2(e)
    return 1.0f - 2.0f * __builtin_amdgcn_rcpf(e + 1.0f);
}

__global__ void convert_f32_bf16(const float* __restrict__ src,
                                 u16* __restrict__ dst, int n) {
    int i = blockIdx.x * blockDim.x + threadIdx.x;
    if (i < n) dst[i] = f32_to_bf16(src[i]);
}

// ---------------------------------------------------------------------------
// TDM: DMA a 2D tile (rows x 64 bf16) from global into LDS, inserting 16B of
// padding after every 128B row so LDS rows land at stride 72 elements.
// D# layout per CDNA5 ISA 8.3/8.4 (Group0 128b, Group1 256b), 2D tensor.
// ---------------------------------------------------------------------------
__device__ __forceinline__ void tdm_load_2d(const u16* lds_dst, const u16* gsrc,
                                            unsigned rows, unsigned row_stride_elems) {
    unsigned lds_off = (unsigned)(uintptr_t)(const void*)lds_dst; // flat LDS addr[31:0] = LDS offset
    unsigned long long ga = (unsigned long long)(uintptr_t)(const void*)gsrc;
    v4u g0 = { 1u,                                   // count=1, user descriptor
               lds_off,                              // lds_addr
               (unsigned)(ga & 0xFFFFFFFFu),         // global_addr[31:0]
               (unsigned)((ga >> 32) & 0x1FFFFFFu) | (2u << 30) }; // [56:32] | type=2
    // Group1: data_size=1(2B), pad_enable, pad_interval=4 (32 DW = 128B),
    // pad_amount=3 (4 DW = 16B) -> LDS row stride 144B = 72 elems.
    v8i g1 = { (int)((1u << 16) | (1u << 20) | (4u << 22) | (3u << 25)),
               (int)(1024u << 16),                   // tensor_dim0 = 1024 (>= tile width)
               (int)0xFFFF0000u,                     // tensor_dim1 = 0xFFFF (>= rows)
               (int)(64u << 16),                     // tile_dim0 = 64
               (int)rows,                            // tile_dim1 = rows, tile_dim2 = 0
               (int)row_stride_elems,                // tensor_dim0_stride[31:0]
               0, 0 };
    v4i z4 = { 0, 0, 0, 0 };
#if __clang_major__ >= 23
    v8i z8 = { 0, 0, 0, 0, 0, 0, 0, 0 };
    __builtin_amdgcn_tensor_load_to_lds(g0, g1, z4, z4, z8, 0);  // 6-arg toolchain
#else
    __builtin_amdgcn_tensor_load_to_lds(g0, g1, z4, z4, 0);      // ROCm 7.2 (clang-22)
#endif
}

// Assemble a 16x32 bf16 WMMA fragment from an LDS tile with row stride 72.
// Each lane: two contiguous 16B runs; lanes 0-15 K={kl..+7, kl+16..+23},
// lanes 16-31 get +8 (folded into kb by caller).
__device__ __forceinline__ v16bf load_frag(const u16* lds, int row, int kb) {
    union { uint4 u[2]; v16bf v; } f;
    f.u[0] = *(const uint4*)(lds + row * 72 + kb);
    f.u[1] = *(const uint4*)(lds + row * 72 + kb + 16);
    return f.v;
}

__device__ __forceinline__ v8f wmma_bf16(v16bf a, v16bf b, v8f c) {
    return __builtin_amdgcn_wmma_f32_16x16x32_bf16(false, a, false, b,
                                                   (short)0, c, false, false);
}

// One 64-wide K chunk: 2x2 register-blocked micro-tile per wave (32m x 32i),
// 8 WMMAs per chunk from 4 A-frags + 4 B-frags (1.0 frag-loads per WMMA).
__device__ __forceinline__ void mm_chunk(const u16* As, const u16* Bs,
                                         int m_off, int i_off, int frow, int fkb,
                                         v8f& c00, v8f& c01, v8f& c10, v8f& c11) {
#pragma unroll
    for (int ks = 0; ks < 2; ++ks) {
        const int kl = ks * 32 + fkb;
        v16bf a0 = load_frag(As, m_off + frow, kl);
        v16bf a1 = load_frag(As, m_off + 16 + frow, kl);
        v16bf b0 = load_frag(Bs, i_off + frow, kl);
        v16bf b1 = load_frag(Bs, i_off + 16 + frow, kl);
        c00 = wmma_bf16(a0, b0, c00);
        c01 = wmma_bf16(a0, b1, c01);
        c10 = wmma_bf16(a1, b0, c10);
        c11 = wmma_bf16(a1, b1, c11);
    }
}

// One recurrence step: Snew = tanh(Sprev @ Wres^T + X_t @ Win^T)
// grid = (H/128, N/64, B), block = 256 threads (8 waves), 64m x 128i tile.
// TDM double-buffers the A/B tiles into LDS while WMMAs consume the other.
__global__ __launch_bounds__(256) void esn_step(
    const u16* __restrict__ Sprev_bf, u16* __restrict__ Snew_bf,
    float* __restrict__ Snew_f32,
    const u16* __restrict__ Wres_bf, const u16* __restrict__ Win_bf,
    const u16* __restrict__ X_bf, int t, int first) {
    __shared__ u16 As[2][64 * 72];    // S / X tile    [m][k], 9 KB each
    __shared__ u16 Bs[2][128 * 72];   // Wres / Win tile [i][k], 18 KB each

    const int tid  = threadIdx.x;
    const int lane = tid & 31;
    const int w    = tid >> 5;
    const int b    = blockIdx.z;
    const int i0   = blockIdx.x * 128;
    const int n0   = blockIdx.y * 64;
    const int m_off = (w >> 2) * 32;   // 2 wave-rows of 32
    const int i_off = (w & 3) * 32;    // 4 wave-cols of 32
    const int frow = lane & 15;
    const int fkb  = (lane >> 4) * 8;

    v8f c00 = {0.f,0.f,0.f,0.f,0.f,0.f,0.f,0.f};
    v8f c01 = c00, c10 = c00, c11 = c00;

    const u16* Sp = Sprev_bf + (long)b * N_S * H_S;
    const u16* Wr = Wres_bf  + (long)b * H_S * H_S;
    const u16* Wi = Win_bf   + (long)b * H_S * V_S;

    if (!first) {
        const int NC = H_S / 64;
        if (w == 0) {   // wave 0 drives the TDM (EXEC-independent, per-wave DMA)
            tdm_load_2d(As[0], Sp + (long)n0 * H_S, 64, H_S);
            tdm_load_2d(Bs[0], Wr + (long)i0 * H_S, 128, H_S);
            __builtin_amdgcn_s_wait_tensorcnt(0);
        }
        __syncthreads();
        for (int kc = 0; kc < NC; ++kc) {
            const int cur = kc & 1;
            if (w == 0 && kc + 1 < NC) {   // prefetch next chunk into other buffer
                const int k1 = (kc + 1) * 64;
                tdm_load_2d(As[cur ^ 1], Sp + (long)n0 * H_S + k1, 64, H_S);
                tdm_load_2d(Bs[cur ^ 1], Wr + (long)i0 * H_S + k1, 128, H_S);
            }
            mm_chunk(As[cur], Bs[cur], m_off, i_off, frow, fkb, c00, c01, c10, c11);
            if (w == 0) __builtin_amdgcn_s_wait_tensorcnt(0);
            __syncthreads();
        }
    }

    // Input contribution: K = V = 64 (one chunk). X is [N][T][V] bf16.
    if (w == 0) {
        tdm_load_2d(As[0], X_bf + ((long)n0 * T_S + t) * V_S, 64, T_S * V_S);
        tdm_load_2d(Bs[0], Wi + (long)i0 * V_S, 128, V_S);
        __builtin_amdgcn_s_wait_tensorcnt(0);
    }
    __syncthreads();
    mm_chunk(As[0], Bs[0], m_off, i_off, frow, fkb, c00, c01, c10, c11);

    // Epilogue: C layout lane(0-15)=col, VGPR r -> row r (+8 for lanes 16-31)
    const int col   = lane & 15;
    const int rhalf = (lane >> 4) * 8;
    u16*   Sn = Snew_bf  + (long)b * N_S * H_S;
    float* Sf = Snew_f32 + (long)b * N_S * H_S;
#pragma unroll
    for (int r = 0; r < 8; ++r) {
        const int n_a = n0 + m_off + rhalf + r;        // rows of c0x
        const int n_b = n_a + 16;                      // rows of c1x
        const int ia  = i0 + i_off + col;
        const int ib  = ia + 16;
        float v00 = tanh_fast(c00[r]);
        float v01 = tanh_fast(c01[r]);
        float v10 = tanh_fast(c10[r]);
        float v11 = tanh_fast(c11[r]);
        Sn[(long)n_a * H_S + ia] = f32_to_bf16(v00);
        Sn[(long)n_a * H_S + ib] = f32_to_bf16(v01);
        Sn[(long)n_b * H_S + ia] = f32_to_bf16(v10);
        Sn[(long)n_b * H_S + ib] = f32_to_bf16(v11);
        Sf[(long)n_a * H_S + ia] = v00;
        Sf[(long)n_a * H_S + ib] = v01;
        Sf[(long)n_b * H_S + ia] = v10;
        Sf[(long)n_b * H_S + ib] = v11;
    }
}

// Per-timestep readout: one block per sample n; logits[t,n,:] + argmax.
__global__ __launch_bounds__(256) void readout(
    const float* __restrict__ Sf32, const float* __restrict__ Wlin,
    const float* __restrict__ blin, float* __restrict__ logits,
    int* __restrict__ preds, int t) {
    const int n   = blockIdx.x;
    const int tid = threadIdx.x;
    const int F   = B_S * H_S;
    float acc[C_S] = {0.f, 0.f, 0.f, 0.f};
    for (int f = tid; f < F; f += 256) {
        const int b = f >> 10;
        const int i = f & (H_S - 1);
        const float s = Sf32[((long)b * N_S + n) * H_S + i];
#pragma unroll
        for (int c = 0; c < C_S; ++c)
            acc[c] += s * Wlin[((long)t * C_S + c) * F + f];
    }
    __shared__ float red[C_S][256];
#pragma unroll
    for (int c = 0; c < C_S; ++c) red[c][tid] = acc[c];
    __syncthreads();
    for (int s = 128; s > 0; s >>= 1) {
        if (tid < s)
#pragma unroll
            for (int c = 0; c < C_S; ++c) red[c][tid] += red[c][tid + s];
        __syncthreads();
    }
    if (tid == 0) {
        float l[C_S];
        int arg = 0;
#pragma unroll
        for (int c = 0; c < C_S; ++c) {
            l[c] = red[c][0] + blin[t * C_S + c];
            logits[((long)t * N_S + n) * C_S + c] = l[c];
            if (c > 0 && l[c] > l[arg]) arg = c;   // first-occurrence argmax
        }
        preds[(long)t * N_S + n] = arg;
    }
}

extern "C" void kernel_launch(void* const* d_in, const int* in_sizes, int n_in,
                              void* d_out, int out_size, void* d_ws, size_t ws_size,
                              hipStream_t stream) {
    const float* X    = (const float*)d_in[0];  // [N,T,V]
    const float* Wres = (const float*)d_in[1];  // [B,H,H]
    const float* Win  = (const float*)d_in[2];  // [B,H,V]
    const float* Wlin = (const float*)d_in[3];  // [T,C,B*H]
    const float* blin = (const float*)d_in[4];  // [T,C]

    char* ws = (char*)d_ws;
    size_t off = 0;
    auto alloc = [&](size_t bytes) {
        char* p = ws + off;
        off += bytes;
        off = (off + 255) & ~(size_t)255;
        return p;
    };
    u16*   Wres_bf = (u16*)alloc((size_t)B_S * H_S * H_S * 2);
    u16*   Win_bf  = (u16*)alloc((size_t)B_S * H_S * V_S * 2);
    u16*   X_bf    = (u16*)alloc((size_t)N_S * T_S * V_S * 2);
    u16*   S_a     = (u16*)alloc((size_t)B_S * N_S * H_S * 2);
    u16*   S_b     = (u16*)alloc((size_t)B_S * N_S * H_S * 2);
    float* Sf32    = (float*)alloc((size_t)B_S * N_S * H_S * 4);

    float* logits = (float*)d_out;                          // [T,N,C] f32
    int*   preds  = (int*)(logits + (long)T_S * N_S * C_S); // [T,N] i32

    const int nWres = B_S * H_S * H_S;
    const int nWin  = B_S * H_S * V_S;
    const int nX    = N_S * T_S * V_S;
    convert_f32_bf16<<<(nWres + 255) / 256, 256, 0, stream>>>(Wres, Wres_bf, nWres);
    convert_f32_bf16<<<(nWin  + 255) / 256, 256, 0, stream>>>(Win,  Win_bf,  nWin);
    convert_f32_bf16<<<(nX    + 255) / 256, 256, 0, stream>>>(X,    X_bf,    nX);

    dim3 grid(H_S / 128, N_S / 64, B_S);
    for (int t = 0; t < T_S; ++t) {
        u16*       dst = (t & 1) ? S_b : S_a;
        const u16* src = (t & 1) ? S_a : S_b;   // unread when first=1
        esn_step<<<grid, 256, 0, stream>>>(src, dst, Sf32, Wres_bf, Win_bf,
                                           X_bf, t, t == 0 ? 1 : 0);
        readout<<<N_S, 256, 0, stream>>>(Sf32, Wlin, blin, logits, preds, t);
    }
}
